// DecoupledMultiheadAttention_9328668967368
// MI455X (gfx1250) — compile-verified, matrix operand order checked by reading, not run on hardware
//
#include <hip/hip_runtime.h>
#include <stdint.h>

#define E_    1024
#define H_    16
#define HD_   64
#define BSZ_  8
#define T_    1024
#define S_    1024

typedef __attribute__((ext_vector_type(16))) __bf16 v16bf;
typedef __attribute__((ext_vector_type(8)))  float  v8f;
typedef unsigned int v4u __attribute__((ext_vector_type(4)));
typedef int          v8i __attribute__((ext_vector_type(8)));
typedef int          v4i __attribute__((ext_vector_type(4)));

#if defined(__has_builtin)
#if __has_builtin(__builtin_amdgcn_tensor_load_to_lds)
#define HAVE_TDM 1
#endif
#endif

union FragAB { v16bf v; uint4 q[2]; };

__device__ __forceinline__ unsigned short f2bf(float x) {
    unsigned int u = __builtin_bit_cast(unsigned int, x);
    u += 0x7FFFu + ((u >> 16) & 1u);          // round-to-nearest-even
    return (unsigned short)(u >> 16);
}
__device__ __forceinline__ unsigned int pk2(unsigned short a, unsigned short b) {
    return (unsigned int)a | ((unsigned int)b << 16);
}

#ifdef HAVE_TDM
// Issue a TDM 2D tile load (bf16 elements): tile_x elems (dim0) x tile_y rows (dim1),
// row stride = stride_elems, global -> LDS at lds_addr. D# per CDNA5 ISA ch.8.
// amdgpu-toolchain (clang-23) 6-arg form: (g0, g1, g2, g3, g_extra, cpol).
__device__ __forceinline__ void tdm_load_2d_bf16(unsigned lds_addr, const void* gaddr,
                                                 unsigned tile_x, unsigned tile_y,
                                                 unsigned stride_elems) {
    unsigned long long ga = (unsigned long long)(uintptr_t)gaddr;
    const unsigned td0 = 1u << 20, td1 = 1u << 20;     // generous tensor dims (no OOB)
    v4u g0;
    g0.x = 1u;                                         // count=1, is_restore=0
    g0.y = lds_addr;                                   // LDS byte address
    g0.z = (unsigned)(ga & 0xFFFFFFFFu);               // global_addr[31:0]  (bits 95:64)
    g0.w = (unsigned)((ga >> 32) & 0x01FFFFFFu)        // global_addr[56:32] (bits 120:96)
         | (2u << 30);                                 // type=2 ("image")
    v8i g1;
    g1[0] = (int)(1u << 16);                           // wg_mask=0, data_size=1 (2 bytes)
    g1[1] = (int)((td0 & 0xFFFFu) << 16);              // abar_addr=0 | tensor_dim0[15:0]
    g1[2] = (int)((td0 >> 16) | ((td1 & 0xFFFFu) << 16)); // tensor_dim0[31:16] | tensor_dim1[15:0]
    g1[3] = (int)((td1 >> 16) | (tile_x << 16));       // tensor_dim1[31:16] | tile_dim0
    g1[4] = (int)(tile_y);                             // tile_dim1 | tile_dim2=0
    g1[5] = (int)(stride_elems);                       // tensor_dim0_stride[31:0]
    g1[6] = 0;                                         // stride[47:32]=0 | dim1_stride lo
    g1[7] = 0;
    v4i z4 = {0, 0, 0, 0};                             // 2D: groups 2/3 unused
    v8i z8 = {0, 0, 0, 0, 0, 0, 0, 0};
    __builtin_amdgcn_tensor_load_to_lds(g0, g1, z4, z4, z8, 0);
}
#endif

// ---------------------------------------------------------------- weights f32 -> bf16
__global__ __launch_bounds__(256) void cvt_kernel(const float* __restrict__ src,
                                                  unsigned short* __restrict__ dst, int n4) {
    int i = blockIdx.x * 256 + threadIdx.x;
    if (i < n4) {
        float4 v = ((const float4*)src)[i];
        uint2 o;
        o.x = pk2(f2bf(v.x), f2bf(v.y));
        o.y = pk2(f2bf(v.z), f2bf(v.w));
        ((uint2*)dst)[i] = o;
    }
}

// ---------------------------------------------------------------- generic projection GEMM
// Y(8192 x 1024) = X(8192 x 1024) @ W^T + b ; 64x64 tile per block, 4 waves.
// B (bf16 weight) tile is staged via the Tensor Data Mover (async, TENSORcnt),
// overlapping the DMA with the A-tile f32->bf16 conversion done on the VALU.
// mode: 0=Q (scale+bf16,(B,H,T,hd)) 1=K (bf16,(B,H,S,hd)) 2=V (bf16 T,(B,H,hd,S)) 3=O (f32)
__global__ __launch_bounds__(128) void gemm_kernel(const float* __restrict__ X,
                                                   const unsigned short* __restrict__ Wb,
                                                   const float* __restrict__ bias,
                                                   unsigned short* __restrict__ obf,
                                                   float* __restrict__ of32,
                                                   int mode) {
    __shared__ unsigned short lA[64][32];   // 64 M-rows x 32 K, bf16
    __shared__ unsigned short lB[64][32];   // 64 N-rows x 32 K, bf16

    const int tid = threadIdx.x;
    const int w   = tid >> 5, l = tid & 31;
    const int lg  = l >> 4,  ln = l & 15;
    const int bm  = blockIdx.x, bn = blockIdx.y;
    const int srow = tid >> 1;
    const int scol = (tid & 1) * 16;

#ifdef HAVE_TDM
    const unsigned ldsB_addr = (unsigned)(size_t)(&lB[0][0]);  // low 32 bits = LDS offset
#endif

    v8f acc[4] = {};

    for (int kk = 0; kk < E_; kk += 32) {
        // ---- stage B tile: async TDM copy (wave 0 issues; EXEC-independent)
#ifdef HAVE_TDM
        if (tid < 32) {
            const unsigned short* gsrc = Wb + (size_t)(bn * 64) * E_ + kk;
            tdm_load_2d_bf16(ldsB_addr, gsrc, /*tile_x=*/32, /*tile_y=*/64, /*stride=*/E_);
        }
#else
        {
            const unsigned short* src = Wb + (size_t)(bn * 64 + srow) * E_ + kk + scol;
            uint4* dst = (uint4*)&lB[srow][scol];
            dst[0] = *(const uint4*)(src);
            dst[1] = *(const uint4*)(src + 8);
        }
#endif
        // ---- stage A tile: f32 global -> bf16 LDS (16 elems/thread), overlaps the DMA
        {
            const float* src = X + (size_t)(bm * 64 + srow) * E_ + kk + scol;
            unsigned int p[8];
#pragma unroll
            for (int i = 0; i < 4; ++i) {
                float4 v = *(const float4*)(src + i * 4);
                p[2 * i]     = pk2(f2bf(v.x), f2bf(v.y));
                p[2 * i + 1] = pk2(f2bf(v.z), f2bf(v.w));
            }
            uint4* dst = (uint4*)&lA[srow][scol];
            dst[0] = make_uint4(p[0], p[1], p[2], p[3]);
            dst[1] = make_uint4(p[4], p[5], p[6], p[7]);
        }
#ifdef HAVE_TDM
        if (tid < 32) __builtin_amdgcn_s_wait_tensorcnt(0);
#endif
        __syncthreads();

        // A fragment: lane ln = row M, lane-group lg selects K {8lg..8lg+7, 16+8lg..}
        FragAB fa;
        fa.q[0] = *(const uint4*)&lA[w * 16 + ln][8 * lg];
        fa.q[1] = *(const uint4*)&lA[w * 16 + ln][16 + 8 * lg];
#pragma unroll
        for (int nt = 0; nt < 4; ++nt) {
            FragAB fb;
            fb.q[0] = *(const uint4*)&lB[nt * 16 + ln][8 * lg];
            fb.q[1] = *(const uint4*)&lB[nt * 16 + ln][16 + 8 * lg];
            acc[nt] = __builtin_amdgcn_wmma_f32_16x16x32_bf16(false, fa.v, false, fb.v,
                                                              (short)0, acc[nt], false, false);
        }
        __syncthreads();
    }

    // epilogue: D layout -> lane l holds N=l%16, M=(l/16)*8+r in acc element r
#pragma unroll
    for (int nt = 0; nt < 4; ++nt) {
        const int   n_g = bn * 64 + nt * 16 + ln;
        const float bvv = bias[n_g];
#pragma unroll
        for (int r = 0; r < 8; ++r) {
            const int m_g = bm * 64 + w * 16 + lg * 8 + r;
            float val = acc[nt][r] + bvv;
            if (mode == 0) {            // Q: rows = b*T+t, scale 1/sqrt(64)
                val *= 0.125f;
                int bb = m_g >> 10, t = m_g & 1023;
                int hh = n_g >> 6, d = n_g & 63;
                obf[((size_t)(bb * H_ + hh) * T_ + t) * HD_ + d] = f2bf(val);
            } else if (mode == 1) {     // K: rows = s*B+b
                int s = m_g >> 3, bb = m_g & 7;
                int hh = n_g >> 6, d = n_g & 63;
                obf[((size_t)(bb * H_ + hh) * S_ + s) * HD_ + d] = f2bf(val);
            } else if (mode == 2) {     // V transposed: (B,H,hd,S)
                int s = m_g >> 3, bb = m_g & 7;
                int hh = n_g >> 6, d = n_g & 63;
                obf[((size_t)(bb * H_ + hh) * HD_ + d) * S_ + s] = f2bf(val);
            } else {                    // O: rows = t*B+b, direct f32 output (T,B,E)
                of32[(size_t)m_g * E_ + n_g] = val;
            }
        }
    }
}

// ---------------------------------------------------------------- attention core
// One block per (b, 16-row T tile); loops all 16 heads (atomic-free head-mean).
__global__ __launch_bounds__(128) void attn_kernel(const unsigned short* __restrict__ qb,
                                                   const unsigned short* __restrict__ kb,
                                                   const unsigned short* __restrict__ vtb,
                                                   const unsigned char* __restrict__ mask,
                                                   float* __restrict__ ctx,
                                                   float* __restrict__ wacc,
                                                   float* __restrict__ wout) {
    __shared__ float        sc[16][S_];        // 64 KB f32 scores / probs
    __shared__ unsigned int wpk[16][S_ / 2];   // 32 KB bf16-packed weights
    __shared__ float        rmx[16][8];
    __shared__ float        rsm[16][8];

    const int tid = threadIdx.x;
    const int w = tid >> 5, l = tid & 31;
    const int lg = l >> 4, ln = l & 15;
    const int b  = blockIdx.x >> 6;
    const int t0 = (blockIdx.x & 63) << 4;

    const int row = tid >> 3;       // softmax row ownership: 8 threads per row
    const int u   = tid & 7;
    const int c0  = u << 7;         // 128 contiguous columns each

    const unsigned char* mrow = mask + (size_t)b * S_;

    for (int h = 0; h < H_; ++h) {
        const size_t bh = (size_t)(b * H_ + h);

        // ---- Q A-fragments (16 x 64, two 16x32 chunks) straight from global
        const unsigned short* qrow = qb + (bh * T_ + t0 + ln) * HD_;
        FragAB aq0, aq1;
        aq0.q[0] = *(const uint4*)(qrow + 8 * lg);
        aq0.q[1] = *(const uint4*)(qrow + 16 + 8 * lg);
        aq1.q[0] = *(const uint4*)(qrow + 32 + 8 * lg);
        aq1.q[1] = *(const uint4*)(qrow + 48 + 8 * lg);

        // ---- scores = q @ k^T : waves split S into 16-wide tiles
        for (int st = w; st < S_ / 16; st += 4) {
            const unsigned short* krow = kb + (bh * S_ + st * 16 + ln) * HD_;
            __builtin_prefetch(krow + 4 * 16 * HD_, 0, 1);   // next tile for this wave
            FragAB bk0, bk1;
            bk0.q[0] = *(const uint4*)(krow + 8 * lg);
            bk0.q[1] = *(const uint4*)(krow + 16 + 8 * lg);
            bk1.q[0] = *(const uint4*)(krow + 32 + 8 * lg);
            bk1.q[1] = *(const uint4*)(krow + 48 + 8 * lg);
            v8f acc = {};
            acc = __builtin_amdgcn_wmma_f32_16x16x32_bf16(false, aq0.v, false, bk0.v,
                                                          (short)0, acc, false, false);
            acc = __builtin_amdgcn_wmma_f32_16x16x32_bf16(false, aq1.v, false, bk1.v,
                                                          (short)0, acc, false, false);
#pragma unroll
            for (int r = 0; r < 8; ++r)
                sc[lg * 8 + r][st * 16 + ln] = acc[r];
        }
        __syncthreads();

        // ---- masked softmax (f32), row-parallel
        float mx = -3.0e38f;
        for (int c = c0; c < c0 + 128; ++c) {
            float x = mrow[c] ? -1.0e30f : sc[row][c];
            mx = fmaxf(mx, x);
        }
        rmx[row][u] = mx;
        __syncthreads();
#pragma unroll
        for (int i = 0; i < 8; ++i) mx = fmaxf(mx, rmx[row][i]);

        float sum = 0.f;
        for (int c = c0; c < c0 + 128; ++c) {
            float p = mrow[c] ? 0.f : __expf(sc[row][c] - mx);
            sc[row][c] = p;
            sum += p;
        }
        rsm[row][u] = sum;
        __syncthreads();
        float tot = 0.f;
#pragma unroll
        for (int i = 0; i < 8; ++i) tot += rsm[row][i];
        const float rinv = 1.0f / tot;

        // ---- accumulate head-mean into ws + pack normalized weights to bf16
        float* wr = wacc + ((size_t)(b * T_ + t0 + row)) * S_;
        for (int c = c0; c < c0 + 128; c += 2) {
            float p0 = sc[row][c] * rinv;
            float p1 = sc[row][c + 1] * rinv;
            float2* wp = (float2*)(wr + c);
            float2 cur;
            if (h == 0) { cur.x = 0.f; cur.y = 0.f; } else { cur = *wp; }
            cur.x += p0 * (1.0f / H_);
            cur.y += p1 * (1.0f / H_);
            *wp = cur;
            wpk[row][c >> 1] = pk2(f2bf(p0), f2bf(p1));
        }
        __syncthreads();

        // ---- ctx = weights @ V : each wave owns 16 of the 64 hd columns, K=S loop
        v8f oacc = {};
        const unsigned short* wprow = (const unsigned short*)(&wpk[ln][0]);
        const unsigned short* vrow  = vtb + (bh * HD_ + w * 16 + ln) * S_;
        for (int s0 = 0; s0 < S_; s0 += 32) {
            __builtin_prefetch(vrow + s0 + 256, 0, 1);
            FragAB aw, bv;
            aw.q[0] = *(const uint4*)(wprow + s0 + 8 * lg);
            aw.q[1] = *(const uint4*)(wprow + s0 + 16 + 8 * lg);
            bv.q[0] = *(const uint4*)(vrow + s0 + 8 * lg);
            bv.q[1] = *(const uint4*)(vrow + s0 + 16 + 8 * lg);
            oacc = __builtin_amdgcn_wmma_f32_16x16x32_bf16(false, aw.v, false, bv.v,
                                                           (short)0, oacc, false, false);
        }
#pragma unroll
        for (int r = 0; r < 8; ++r) {
            const int tl = lg * 8 + r;
            ctx[((size_t)(t0 + tl) * BSZ_ + b) * E_ + h * HD_ + w * 16 + ln] = oacc[r];
        }
        __syncthreads();
    }

    // ---- emit mean weights as (S, T, B)
    {
        const float* wr = wacc + ((size_t)(b * T_ + t0 + row)) * S_;
        const int tg = t0 + row;
        for (int c = c0; c < c0 + 128; ++c)
            wout[((size_t)c * T_ + tg) * BSZ_ + b] = wr[c];
    }
}

// ---------------------------------------------------------------- launcher
extern "C" void kernel_launch(void* const* d_in, const int* in_sizes, int n_in,
                              void* d_out, int out_size, void* d_ws, size_t ws_size,
                              hipStream_t stream) {
    (void)in_sizes; (void)n_in; (void)out_size; (void)ws_size;

    const float*         dec  = (const float*)d_in[0];          // (B,T,E)
    const float*         srch = (const float*)d_in[1];          // (S,B,E)
    const unsigned char* mask = (const unsigned char*)d_in[2];  // (B,S) bool
    const float* Wq = (const float*)d_in[3];
    const float* bq = (const float*)d_in[4];
    const float* Wk = (const float*)d_in[5];
    const float* bk = (const float*)d_in[6];
    const float* Wv = (const float*)d_in[7];
    const float* bv = (const float*)d_in[8];
    const float* Wo = (const float*)d_in[9];
    const float* bo = (const float*)d_in[10];

    unsigned short* Wqb  = (unsigned short*)d_ws;               // 4x E*E bf16
    unsigned short* Wkb  = Wqb + (size_t)E_ * E_;
    unsigned short* Wvb  = Wkb + (size_t)E_ * E_;
    unsigned short* Wob  = Wvb + (size_t)E_ * E_;
    unsigned short* qbuf = Wob + (size_t)E_ * E_;               // (B,H,T,hd) bf16
    unsigned short* kbuf = qbuf + (size_t)BSZ_ * E_ * T_;       // (B,H,S,hd) bf16
    unsigned short* vtbf = kbuf + (size_t)BSZ_ * E_ * S_;       // (B,H,hd,S) bf16
    float*          ctx  = (float*)(vtbf + (size_t)BSZ_ * E_ * S_); // (T*B, E) f32
    float*          wacc = ctx + (size_t)T_ * BSZ_ * E_;        // (B,T,S) f32

    float* attn_out = (float*)d_out;                            // (T,B,E)
    float* w_out    = attn_out + (size_t)T_ * BSZ_ * E_;        // (S,T,B)

    const int n4 = E_ * E_ / 4;
    cvt_kernel<<<dim3(n4 / 256), dim3(256), 0, stream>>>(Wq, Wqb, n4);
    cvt_kernel<<<dim3(n4 / 256), dim3(256), 0, stream>>>(Wk, Wkb, n4);
    cvt_kernel<<<dim3(n4 / 256), dim3(256), 0, stream>>>(Wv, Wvb, n4);
    cvt_kernel<<<dim3(n4 / 256), dim3(256), 0, stream>>>(Wo, Wob, n4);

    dim3 gg(128, 16), gb(128);
    gemm_kernel<<<gg, gb, 0, stream>>>(dec,  Wqb, bq, qbuf, nullptr, 0);
    gemm_kernel<<<gg, gb, 0, stream>>>(srch, Wkb, bk, kbuf, nullptr, 1);
    gemm_kernel<<<gg, gb, 0, stream>>>(srch, Wvb, bv, vtbf, nullptr, 2);

    attn_kernel<<<dim3(BSZ_ * (T_ / 16)), dim3(128), 0, stream>>>(qbuf, kbuf, vtbf, mask,
                                                                  ctx, wacc, w_out);

    gemm_kernel<<<gg, gb, 0, stream>>>(ctx, Wob, bo, nullptr, attn_out, 3);
}
